// BlindPnPNeuralSolver_80333068304780
// MI455X (gfx1250) — matrix operand order repaired and checked by reading, not compiled, not run on hardware
//
#include <hip/hip_runtime.h>
#include <hip/hip_bf16.h>
#include <math.h>

typedef __attribute__((ext_vector_type(16))) _Float16 v16h;
typedef __attribute__((ext_vector_type(8)))  _Float16 v8h;
typedef __attribute__((ext_vector_type(8)))  float    v8f;

#define NPTS 4096
#define DIM  512
#define KNN  16
#define NROW (NPTS*KNN)   /* 65536 */

// ======================================================================
// Small prep kernels
// ======================================================================
__global__ void inv3x3_kernel(const float* __restrict__ Km, float* __restrict__ inv) {
    if (threadIdx.x != 0 || blockIdx.x != 0) return;
    float a = Km[0], b = Km[1], c = Km[2];
    float d = Km[3], e = Km[4], f = Km[5];
    float g = Km[6], h = Km[7], i = Km[8];
    float det = a*(e*i - f*h) - b*(d*i - f*g) + c*(d*h - e*g);
    float r = 1.0f / det;
    inv[0] = (e*i - f*h)*r; inv[1] = (c*h - b*i)*r; inv[2] = (b*f - c*e)*r;
    inv[3] = (f*g - d*i)*r; inv[4] = (a*i - c*g)*r; inv[5] = (c*d - a*f)*r;
    inv[6] = (d*h - e*g)*r; inv[7] = (b*g - a*h)*r; inv[8] = (a*e - b*d)*r;
}

__global__ __launch_bounds__(256) void prep_points_kernel(
    const float* __restrict__ pix, const float* __restrict__ pts,
    const float* __restrict__ inv, float* __restrict__ nbv2, float* __restrict__ nbv3) {
    int i = blockIdx.x * blockDim.x + threadIdx.x;
    if (i >= NPTS) return;
    float t0 = pix[i*2+1], t1 = pix[i*2+0], t2 = 1.0f;
    float b0 = t0*inv[0] + t1*inv[1] + t2*inv[2];
    float b1 = t0*inv[3] + t1*inv[4] + t2*inv[5];
    float b2 = t0*inv[6] + t1*inv[7] + t2*inv[8];
    float n = fmaxf(sqrtf(b0*b0 + b1*b1 + b2*b2), 1e-12f);
    nbv2[i*3+0] = b0/n; nbv2[i*3+1] = b1/n; nbv2[i*3+2] = b2/n;
    float pz = pts[i*3+2];
    float q0 = pts[i*3+0]/pz, q1 = pts[i*3+1]/pz, q2 = 1.0f;
    float m = fmaxf(sqrtf(q0*q0 + q1*q1 + q2*q2), 1e-12f);
    nbv3[i*3+0] = q0/m; nbv3[i*3+1] = q1/m; nbv3[i*3+2] = q2/m;
}

__global__ __launch_bounds__(256) void cvt_f32_f16_kernel(
    const float* __restrict__ src, _Float16* __restrict__ dst, int n) {
    int i = blockIdx.x * blockDim.x + threadIdx.x;
    if (i < n) dst[i] = (_Float16)src[i];
}

__global__ __launch_bounds__(256) void fea_init_kernel(
    const float* __restrict__ nbv, const float* __restrict__ w,
    const float* __restrict__ b, float* __restrict__ fea) {
    int tid = blockIdx.x * blockDim.x + threadIdx.x;
    if (tid >= NPTS*DIM) return;
    int n = tid & (DIM-1);
    int i = tid >> 9;
    fea[tid] = nbv[i*3+0]*w[n*3+0] + nbv[i*3+1]*w[n*3+1] + nbv[i*3+2]*w[n*3+2] + b[n];
}

// ======================================================================
// KNN: all xyz in LDS (48KB << 320KB/WGP), per-thread insertion top-16
// ======================================================================
__global__ __launch_bounds__(256) void knn_kernel(
    const float* __restrict__ xyz, int* __restrict__ idx) {
    __shared__ float sx[NPTS], sy[NPTS], sz[NPTS];
    for (int i = threadIdx.x; i < NPTS; i += blockDim.x) {
        sx[i] = xyz[i*3+0]; sy[i] = xyz[i*3+1]; sz[i] = xyz[i*3+2];
    }
    __syncthreads();
    int q = blockIdx.x * blockDim.x + threadIdx.x;
    if (q >= NPTS) return;
    float qx = sx[q], qy = sy[q], qz = sz[q];
    float bd[KNN]; int bi[KNN];
    #pragma unroll
    for (int r = 0; r < KNN; ++r) { bd[r] = 3.4e38f; bi[r] = 0; }
    for (int j = 0; j < NPTS; ++j) {
        float dx = qx - sx[j], dy = qy - sy[j], dz = qz - sz[j];
        float d = dx*dx + dy*dy + dz*dz;
        if (d < bd[KNN-1]) {
            int p = KNN-1;
            while (p > 0 && bd[p-1] > d) { bd[p] = bd[p-1]; bi[p] = bi[p-1]; --p; }
            bd[p] = d; bi[p] = j;
        }
    }
    #pragma unroll
    for (int r = 0; r < KNN; ++r) idx[q*KNN + r] = bi[r];
}

// ======================================================================
// WMMA GEMM: C[M,512] = act(A[M,512](f16) @ W[512,512](f16)^T + bias)
//            (+resid). Template-specialized epilogue (no branches).
// Block = 256 thr = 8 wave32; block tile 128x128; wave tile 32x64.
// A tile (128x32) double-buffered in LDS via GLOBAL_LOAD_ASYNC_TO_LDS_B128
// (ASYNCcnt / s_wait_asynccnt), rows padded to 40 halves for bank spread.
// Fragment layouts used here (wave32; A and B share contiguous K order):
//   A 16x32 f16: lane&15=row, 16 contiguous K halves at 16*hi  (2x ds b128)
//   B 32x16 f16: lane&15=col, 16 contiguous K halves at k0+16*hi (1x 32B)
//   C 16x16 f32: lane&15=col, vgpr r -> row r+8*hi
// ======================================================================
#define A_RS 40   /* LDS row stride in halves (80B) */

template<int ACT, int RES, int OUTF, int OUTH>
__global__ __launch_bounds__(256) void gemm_wmma_kernel(
    const _Float16* __restrict__ A, const _Float16* __restrict__ W,
    const float* __restrict__ bias, const float* __restrict__ resid,
    float* __restrict__ Cf, _Float16* __restrict__ Ch,
    int M, int N, int K) {
    __shared__ _Float16 sA[2][128 * A_RS];   // 2 x 10KB
    const int lane = threadIdx.x & 31;
    const int wave = threadIdx.x >> 5;
    const int wm = wave & 3;             // 4 waves along M (32 rows each)
    const int wn = wave >> 2;            // 2 waves along N (64 cols each)
    const int mrow0 = blockIdx.y * 128;
    const int nbase = blockIdx.x * 128 + wn * 64;
    const int hi = lane >> 4;
    const int lr = lane & 15;

    v8f acc[2][4];
    #pragma unroll
    for (int i = 0; i < 2; ++i)
        #pragma unroll
        for (int j = 0; j < 4; ++j)
            #pragma unroll
            for (int e = 0; e < 8; ++e) acc[i][j][e] = 0.0f;

    const int KT = K >> 5;

    // prologue: async-stage k-slice 0 into LDS buffer 0
    #pragma unroll
    for (int t = 0; t < 2; ++t) {
        int c = threadIdx.x + t * 256;       // 512 x 16B chunks = 128 rows x 64B
        int row = c >> 2, part = c & 3;
        unsigned long long ga =
            (unsigned long long)(uintptr_t)(A + (size_t)(mrow0 + row) * K + part * 8);
        unsigned lo = (unsigned)(uintptr_t)&sA[0][row * A_RS + part * 8];
        asm volatile("global_load_async_to_lds_b128 %0, %1, off"
                     :: "v"(lo), "v"(ga) : "memory");
    }

    for (int kt = 0; kt < KT; ++kt) {
        const int p = kt & 1;
        const int k0 = kt << 5;
        asm volatile("s_wait_asynccnt 0x0" ::: "memory");
        __syncthreads();                      // buffer p ready for all waves
        if (kt + 1 < KT) {                    // stage next slice into buffer p^1
            const int k0n = k0 + 32;
            #pragma unroll
            for (int t = 0; t < 2; ++t) {
                int c = threadIdx.x + t * 256;
                int row = c >> 2, part = c & 3;
                unsigned long long ga =
                    (unsigned long long)(uintptr_t)(A + (size_t)(mrow0 + row) * K + k0n + part * 8);
                unsigned lo = (unsigned)(uintptr_t)&sA[p ^ 1][row * A_RS + part * 8];
                asm volatile("global_load_async_to_lds_b128 %0, %1, off"
                             :: "v"(lo), "v"(ga) : "memory");
            }
            __builtin_prefetch(W + (size_t)(nbase + lr) * K + k0n, 0, 3);
        }
        // A fragments from LDS
        v16h a[2];
        #pragma unroll
        for (int i = 0; i < 2; ++i) {
            const _Float16* lp = &sA[p][(wm * 32 + i * 16 + lr) * A_RS + 16 * hi];
            v8h a0 = *(const v8h*)lp;         // K = 16*hi + [0..7]
            v8h a1 = *(const v8h*)(lp + 8);   // K = 16*hi + 8 + [0..7]
            #pragma unroll
            for (int e = 0; e < 8; ++e) { a[i][e] = a0[e]; a[i][e + 8] = a1[e]; }
        }
        // B fragments direct from global (W is L2-resident)
        v16h b[4];
        #pragma unroll
        for (int j = 0; j < 4; ++j)
            b[j] = *(const v16h*)(W + (size_t)(nbase + j * 16 + lr) * K + k0 + 16 * hi);
        #pragma unroll
        for (int i = 0; i < 2; ++i)
            #pragma unroll
            for (int j = 0; j < 4; ++j)
                acc[i][j] = __builtin_amdgcn_wmma_f32_16x16x32_f16(
                    false, a[i], false, b[j], (short)0, acc[i][j], false, false);
    }

    const int mbase = mrow0 + wm * 32;
    #pragma unroll
    for (int j = 0; j < 4; ++j) {
        int col = nbase + j * 16 + lr;
        float bv = bias[col];
        #pragma unroll
        for (int i = 0; i < 2; ++i) {
            #pragma unroll
            for (int r = 0; r < 8; ++r) {
                int row = mbase + i * 16 + r + 8 * hi;
                float v = acc[i][j][r] + bv;
                if constexpr (ACT)  v = fmaxf(v, 0.0f);
                size_t o = (size_t)row * N + col;
                if constexpr (RES)  v += resid[o];
                if constexpr (OUTF) Cf[o] = v;
                if constexpr (OUTH) Ch[o] = (_Float16)v;
            }
        }
    }
}

// ======================================================================
// Fused elementwise kernels for the attention block
// ======================================================================
__global__ __launch_bounds__(256) void h1_kernel(
    const float* __restrict__ xyz, const int* __restrict__ idx,
    const float* __restrict__ p1w, const float* __restrict__ p1b,
    _Float16* __restrict__ h1) {
    int tid = blockIdx.x * blockDim.x + threadIdx.x;
    if (tid >= NROW*DIM) return;
    int n = tid & (DIM-1);
    int row = tid >> 9;
    int i = row >> 4;
    int g = idx[row];
    float rx = xyz[i*3+0] - xyz[g*3+0];
    float ry = xyz[i*3+1] - xyz[g*3+1];
    float rz = xyz[i*3+2] - xyz[g*3+2];
    float v = rx*p1w[n*3+0] + ry*p1w[n*3+1] + rz*p1w[n*3+2] + p1b[n];
    h1[tid] = (_Float16)fmaxf(v, 0.0f);
}

__global__ __launch_bounds__(256) void t_kernel(
    const float* __restrict__ q, const float* __restrict__ k,
    const float* __restrict__ pe, const int* __restrict__ idx,
    _Float16* __restrict__ t) {
    int tid = blockIdx.x * blockDim.x + threadIdx.x;
    if (tid >= NROW*DIM) return;
    int n = tid & (DIM-1);
    int row = tid >> 9;
    int i = row >> 4;
    int g = idx[row];
    t[tid] = (_Float16)(q[(size_t)i*DIM + n] - k[(size_t)g*DIM + n] + pe[tid]);
}

__global__ __launch_bounds__(256) void softmax_agg_kernel(
    const float* __restrict__ logits, const float* __restrict__ v,
    const float* __restrict__ pe, const int* __restrict__ idx,
    _Float16* __restrict__ aggh) {
    int tid = blockIdx.x * blockDim.x + threadIdx.x;
    if (tid >= NPTS*DIM) return;
    int n = tid & (DIM-1);
    int i = tid >> 9;
    int base = i * KNN;
    float l[KNN];
    float mx = -3.4e38f;
    #pragma unroll
    for (int j = 0; j < KNN; ++j) {
        l[j] = logits[(size_t)(base + j)*DIM + n];
        mx = fmaxf(mx, l[j]);
    }
    float s = 0.0f;
    #pragma unroll
    for (int j = 0; j < KNN; ++j) { l[j] = __expf(l[j] - mx); s += l[j]; }
    float inv = 1.0f / s;
    float acc = 0.0f;
    #pragma unroll
    for (int j = 0; j < KNN; ++j) {
        int g = idx[base + j];
        acc += (l[j] * inv) * (v[(size_t)g*DIM + n] + pe[(size_t)(base + j)*DIM + n]);
    }
    aggh[tid] = (_Float16)acc;
}

// ======================================================================
// Pooling / head / output
// ======================================================================
__global__ void mean_kernel(const float* __restrict__ fea2,
                            const float* __restrict__ fea3,
                            float* __restrict__ posefea) {
    int c = blockIdx.x * blockDim.x + threadIdx.x;
    if (c >= 2*DIM) return;
    const float* src = (c < DIM) ? fea2 : fea3;
    int col = c & (DIM-1);
    float s = 0.0f;
    for (int i = 0; i < NPTS; ++i) s += src[(size_t)i*DIM + col];
    posefea[c] = s * (1.0f / NPTS);
}

__global__ void pose_kernel(const float* __restrict__ hw, const float* __restrict__ hb,
                            const float* __restrict__ posefea, float* __restrict__ RT) {
    if (threadIdx.x != 0 || blockIdx.x != 0) return;
    float pose[6];
    for (int r = 0; r < 6; ++r) {
        float s = hb[r];
        for (int c = 0; c < 2*DIM; ++c) s += hw[r*2*DIM + c] * posefea[c];
        pose[r] = s;
    }
    float a0 = pose[0], a1 = pose[1], a2 = pose[2];
    float ang = sqrtf(a0*a0 + a1*a1 + a2*a2 + 1e-12f);
    float x = a0/ang, y = a1/ang, z = a2/ang;
    float c = cosf(ang), s = sinf(ang), t = 1.0f - c;
    RT[0] = c + t*x*x;    RT[1] = -s*z + t*x*y; RT[2] = s*y + t*x*z;
    RT[3] = s*z + t*y*x;  RT[4] = c + t*y*y;    RT[5] = -s*x + t*y*z;
    RT[6] = -s*y + t*z*x; RT[7] = s*x + t*z*y;  RT[8] = c + t*z*z;
    RT[9] = pose[3]; RT[10] = pose[4]; RT[11] = pose[5];
}

__global__ __launch_bounds__(256) void final_kernel(
    const float* __restrict__ nbv2, const float* __restrict__ RT,
    float* __restrict__ out) {
    int i = blockIdx.x * blockDim.x + threadIdx.x;
    if (i >= NPTS) return;
    float b0 = nbv2[i*3+0], b1 = nbv2[i*3+1], b2 = nbv2[i*3+2];
    float v0 = b0*RT[0] + b1*RT[3] + b2*RT[6] + RT[9];
    float v1 = b0*RT[1] + b1*RT[4] + b2*RT[7] + RT[10];
    float v2 = b0*RT[2] + b1*RT[5] + b2*RT[8] + RT[11];
    float n = fmaxf(sqrtf(v0*v0 + v1*v1 + v2*v2), 1e-12f);
    out[0*NPTS + i] = v0/n; out[1*NPTS + i] = v1/n; out[2*NPTS + i] = v2/n;
}

// ======================================================================
// Host-side orchestration
// ======================================================================
namespace {

struct BlockW16 {
    _Float16 *qw, *kw, *vw, *p2w, *a1w, *a2w, *ow;
};

struct Scratch {
    float *invK, *RT, *posefea;
    float *nbv2, *nbv3;
    int   *idx2, *idx3;
    float *fea2, *fea3;
    _Float16 *feah, *aggh;
    float *q, *k, *v;
    _Float16 *h1t;       // h1, later aliased as t (pe GEMM consumed h1 first)
    _Float16 *h2;
    float *pe, *logits;
    BlockW16 w16[2];
};

inline void run_pt_block(const float* const* bp, const BlockW16& w,
                         const float* xyz, const int* idx, float* fea,
                         const Scratch& S, hipStream_t stream) {
    const int EW_PT  = (NPTS*DIM) / 256;
    const int EW_ROW = (NROW*DIM) / 256;
    dim3 g4k(DIM/128, NPTS/128);   // (4, 32)
    dim3 g64k(DIM/128, NROW/128);  // (4, 512)

    cvt_f32_f16_kernel<<<EW_PT, 256, 0, stream>>>(fea, S.feah, NPTS*DIM);
    // q, k, v  (f32 out, no act, no resid)
    gemm_wmma_kernel<0,0,1,0><<<g4k, 256, 0, stream>>>(S.feah, w.qw, bp[1], nullptr, S.q, nullptr, NPTS, DIM, DIM);
    gemm_wmma_kernel<0,0,1,0><<<g4k, 256, 0, stream>>>(S.feah, w.kw, bp[3], nullptr, S.k, nullptr, NPTS, DIM, DIM);
    gemm_wmma_kernel<0,0,1,0><<<g4k, 256, 0, stream>>>(S.feah, w.vw, bp[5], nullptr, S.v, nullptr, NPTS, DIM, DIM);
    // h1 = relu(rel @ p1w^T + p1b)
    h1_kernel<<<EW_ROW, 256, 0, stream>>>(xyz, idx, bp[6], bp[7], S.h1t);
    // pe = h1 @ p2w^T + p2b  (f32)
    gemm_wmma_kernel<0,0,1,0><<<g64k, 256, 0, stream>>>(S.h1t, w.p2w, bp[9], nullptr, S.pe, nullptr, NROW, DIM, DIM);
    // t = q - k[idx] + pe  (f16, reuses h1 buffer)
    t_kernel<<<EW_ROW, 256, 0, stream>>>(S.q, S.k, S.pe, idx, S.h1t);
    // h2 = relu(t @ a1w^T + a1b)  (f16 only)
    gemm_wmma_kernel<1,0,0,1><<<g64k, 256, 0, stream>>>(S.h1t, w.a1w, bp[11], nullptr, nullptr, S.h2, NROW, DIM, DIM);
    // logits = h2 @ a2w^T + a2b  (f32)
    gemm_wmma_kernel<0,0,1,0><<<g64k, 256, 0, stream>>>(S.h2, w.a2w, bp[13], nullptr, S.logits, nullptr, NROW, DIM, DIM);
    // softmax over 16 + aggregate -> aggh (f16)
    softmax_agg_kernel<<<EW_PT, 256, 0, stream>>>(S.logits, S.v, S.pe, idx, S.aggh);
    // fea = aggh @ ow^T + ob + fea   (in-place residual)
    gemm_wmma_kernel<0,1,1,0><<<g4k, 256, 0, stream>>>(S.aggh, w.ow, bp[15], fea, fea, nullptr, NPTS, DIM, DIM);
}

} // namespace

extern "C" void kernel_launch(void* const* d_in, const int* in_sizes, int n_in,
                              void* d_out, int out_size, void* d_ws, size_t ws_size,
                              hipStream_t stream) {
    (void)in_sizes; (void)n_in; (void)out_size; (void)ws_size;
    const float* pix   = (const float*)d_in[0];
    const float* pts   = (const float*)d_in[1];
    const float* intr  = (const float*)d_in[2];
    const float* pre2w = (const float*)d_in[3];
    const float* pre2b = (const float*)d_in[4];
    const float* pre3w = (const float*)d_in[5];
    const float* pre3b = (const float*)d_in[6];
    const float* blk1[16]; const float* blk2[16];
    for (int i = 0; i < 16; ++i) { blk1[i] = (const float*)d_in[7+i]; blk2[i] = (const float*)d_in[23+i]; }
    const float* headw = (const float*)d_in[39];
    const float* headb = (const float*)d_in[40];
    float* out = (float*)d_out;

    char* base = (char*)d_ws;
    size_t off = 0;
    auto alloc = [&](size_t bytes) -> char* {
        char* p = base + off;
        off = (off + bytes + 255) & ~(size_t)255;
        return p;
    };
    Scratch S;
    S.invK    = (float*)alloc(9*4);
    S.RT      = (float*)alloc(12*4);
    S.posefea = (float*)alloc(2*DIM*4);
    S.nbv2    = (float*)alloc((size_t)NPTS*3*4);
    S.nbv3    = (float*)alloc((size_t)NPTS*3*4);
    S.idx2    = (int*)  alloc((size_t)NPTS*KNN*4);
    S.idx3    = (int*)  alloc((size_t)NPTS*KNN*4);
    S.fea2    = (float*)alloc((size_t)NPTS*DIM*4);
    S.fea3    = (float*)alloc((size_t)NPTS*DIM*4);
    S.feah    = (_Float16*)alloc((size_t)NPTS*DIM*2);
    S.aggh    = (_Float16*)alloc((size_t)NPTS*DIM*2);
    S.q       = (float*)alloc((size_t)NPTS*DIM*4);
    S.k       = (float*)alloc((size_t)NPTS*DIM*4);
    S.v       = (float*)alloc((size_t)NPTS*DIM*4);
    S.h1t     = (_Float16*)alloc((size_t)NROW*DIM*2);
    S.h2      = (_Float16*)alloc((size_t)NROW*DIM*2);
    S.pe      = (float*)alloc((size_t)NROW*DIM*4);
    S.logits  = (float*)alloc((size_t)NROW*DIM*4);
    for (int b = 0; b < 2; ++b) {
        S.w16[b].qw  = (_Float16*)alloc((size_t)DIM*DIM*2);
        S.w16[b].kw  = (_Float16*)alloc((size_t)DIM*DIM*2);
        S.w16[b].vw  = (_Float16*)alloc((size_t)DIM*DIM*2);
        S.w16[b].p2w = (_Float16*)alloc((size_t)DIM*DIM*2);
        S.w16[b].a1w = (_Float16*)alloc((size_t)DIM*DIM*2);
        S.w16[b].a2w = (_Float16*)alloc((size_t)DIM*DIM*2);
        S.w16[b].ow  = (_Float16*)alloc((size_t)DIM*DIM*2);
    }

    inv3x3_kernel<<<1, 1, 0, stream>>>(intr, S.invK);
    prep_points_kernel<<<NPTS/256, 256, 0, stream>>>(pix, pts, S.invK, S.nbv2, S.nbv3);
    knn_kernel<<<NPTS/256, 256, 0, stream>>>(S.nbv2, S.idx2);
    knn_kernel<<<NPTS/256, 256, 0, stream>>>(S.nbv3, S.idx3);
    fea_init_kernel<<<(NPTS*DIM)/256, 256, 0, stream>>>(S.nbv2, pre2w, pre2b, S.fea2);
    fea_init_kernel<<<(NPTS*DIM)/256, 256, 0, stream>>>(S.nbv3, pre3w, pre3b, S.fea3);

    const int WIDX[7] = {0, 2, 4, 8, 10, 12, 14};
    const int NW = DIM*DIM;
    for (int b = 0; b < 2; ++b) {
        const float* const* bp = (b == 0) ? blk1 : blk2;
        _Float16* dsts[7] = {S.w16[b].qw, S.w16[b].kw, S.w16[b].vw, S.w16[b].p2w,
                             S.w16[b].a1w, S.w16[b].a2w, S.w16[b].ow};
        for (int wI = 0; wI < 7; ++wI)
            cvt_f32_f16_kernel<<<NW/256, 256, 0, stream>>>(bp[WIDX[wI]], dsts[wI], NW);
    }

    run_pt_block(blk1, S.w16[0], S.nbv2, S.idx2, S.fea2, S, stream);
    run_pt_block(blk1, S.w16[0], S.nbv3, S.idx3, S.fea3, S, stream);
    run_pt_block(blk2, S.w16[1], S.nbv2, S.idx2, S.fea2, S, stream);
    run_pt_block(blk2, S.w16[1], S.nbv3, S.idx3, S.fea3, S, stream);

    mean_kernel<<<(2*DIM)/256, 256, 0, stream>>>(S.fea2, S.fea3, S.posefea);
    pose_kernel<<<1, 1, 0, stream>>>(headw, headb, S.posefea, S.RT);
    final_kernel<<<NPTS/256, 256, 0, stream>>>(S.nbv2, S.RT, out);
}
